// ACSConv_3264175145499
// MI455X (gfx1250) — matrix operand order repaired and checked by reading, not compiled, not run on hardware
//
#include <hip/hip_runtime.h>
#include <hip/hip_bf16.h>

// ---------------------------------------------------------------------------
// Chebyshev graph conv (ACSConv) for gfx1250.
//   T0 = tile(x) ; out = feat(T0) @ W0 + bias
//   T1 = L @ T0  ; out += feat(T1) @ W1
//   Tk = 2 L T_{k-1} - T_{k-2} ; out += feat(Tk) @ Wk   (k = 2..7)
// feat(T)[n, a*64+c] = T[a*N + n, c]   (K-dim = 8 angles * 64 chan = 512)
//
// SpMM: edge-parallel, one edge per wave32 (2 features/lane), f32 atomics.
// GEMM: wave-level WMMA f32_16x16x32_f16, f32 accumulate.
//       A tile streamed global->LDS with GLOBAL_LOAD_ASYNC_TO_LDS_B128
//       (ASYNCcnt + s_wait_asynccnt), f32->f16 conversion in registers.
// ---------------------------------------------------------------------------

typedef __attribute__((ext_vector_type(16))) _Float16 v16h;
typedef __attribute__((ext_vector_type(8)))  _Float16 v8h;
typedef __attribute__((ext_vector_type(8)))  float    v8f;

#define N_ANGLES 8
#define IN_SIZE  64
#define OUT_SIZE 64
#define KDIM     (N_ANGLES * IN_SIZE)   // 512
#define TILE_NODES 128                  // per block (8 waves * 16 rows)

// ---- Tx0 = tile(x, (8,1)) : T0[a*N + n][c] = x[n][c] --------------------
__global__ void tile_kernel(const float4* __restrict__ x4,
                            float4* __restrict__ T4,
                            int N, int totalV4) {
    int idx = blockIdx.x * blockDim.x + threadIdx.x;
    if (idx >= totalV4) return;
    int row  = idx >> 4;          // 16 float4 per 64-float row
    int c4   = idx & 15;
    int node = row % N;
    T4[idx] = x4[node * 16 + c4];
}

// ---- Y = alpha * Y  (alpha==0 -> hard zero; handles poisoned ws) --------
__global__ void scale_kernel(float4* __restrict__ Y, float alpha, int totalV4) {
    int idx = blockIdx.x * blockDim.x + threadIdx.x;
    if (idx >= totalV4) return;
    if (alpha == 0.0f) {
        Y[idx] = make_float4(0.f, 0.f, 0.f, 0.f);
    } else {
        float4 v = Y[idx];
        v.x *= alpha; v.y *= alpha; v.z *= alpha; v.w *= alpha;
        Y[idx] = v;
    }
}

// ---- Y += alpha * (L @ Tin) : one edge per wave32, 2 features/lane ------
__global__ __launch_bounds__(256)
void spmm_kernel(const float* __restrict__ vals,
                 const int*   __restrict__ rows,
                 const int*   __restrict__ cols,
                 const float* __restrict__ Tin,
                 float*       __restrict__ Y,
                 float alpha, int E) {
    int gtid = blockIdx.x * blockDim.x + threadIdx.x;
    int e    = gtid >> 5;
    int lane = gtid & 31;
    if (e >= E) return;
    float v = alpha * vals[e];          // wave-uniform loads
    int   r = rows[e];
    int   c = cols[e];
    const float* tin  = Tin + (size_t)c * 64;  // 256B row, L2-resident gather
    float*       yout = Y   + (size_t)r * 64;
    float f0 = tin[lane];
    float f1 = tin[lane + 32];
    __hip_atomic_fetch_add(&yout[lane],      v * f0,
                           __ATOMIC_RELAXED, __HIP_MEMORY_SCOPE_AGENT);
    __hip_atomic_fetch_add(&yout[lane + 32], v * f1,
                           __ATOMIC_RELAXED, __HIP_MEMORY_SCOPE_AGENT);
}

// ---- out (+)= feat(T) @ Wk [+ bias]  via WMMA f16 -----------------------
// Block: 256 threads = 8 waves; each wave computes a 16x64 output tile.
// Per angle a: async-stage A (128 nodes x 64 f32) and stage W slice
// (64 kk x 64 n, transposed f16) in LDS, then 2 k-steps of 16x16x32 WMMA.
__global__ __launch_bounds__(256)
void gemm_kernel(const float* __restrict__ T,
                 const float* __restrict__ Wk,     // [512][64] row-major f32
                 const float* __restrict__ bias,
                 float*       __restrict__ out,
                 int N, int initOut) {
    __shared__ float    __attribute__((aligned(32))) ldsA32[TILE_NODES * 64]; // [node][c] f32
    __shared__ _Float16 __attribute__((aligned(32))) ldsW[64 * 64];           // [n][kk_local]

    const int tid      = threadIdx.x;
    const int wave     = tid >> 5;
    const int lane     = tid & 31;
    const int nodeBase = blockIdx.x * TILE_NODES;

    // Raw LDS byte offset of the A tile (VDST of async load holds an
    // offset relative to the wave's LDS base; addrspace(3) ptrtoint).
    const unsigned ldsABase =
        (unsigned)(size_t)(__attribute__((address_space(3))) float*)ldsA32;

    v8f acc[4];
#pragma unroll
    for (int t = 0; t < 4; ++t) acc[t] = (v8f){};

    // Per-lane fragment geometry (contiguous k = 16h + s order used here for A and B, wave32):
    // A 16x32: lanes 0-15 -> K 0..15;         lanes 16-31 -> K 16..31
    // B 32x16: lanes 0-15 -> K 0..15;         lanes 16-31 -> K 16..31
    const int m      = lane & 15;
    const int deltaA = (lane < 16) ? 0 : 16;
    const int deltaB = (lane < 16) ? 0 : 16;

    for (int a = 0; a < N_ANGLES; ++a) {
        __syncthreads();   // previous iteration's LDS reads done

        // --- Async-stage A tile: 128 rows x 256B = 2048 x 16B chunks. ---
        // ASYNCcnt-tracked, bypasses VGPRs. Rows past N zero-filled via
        // regular ds stores (disjoint chunks, independent counter).
        const float* srcA = T + ((size_t)a * N + nodeBase) * 64;
        for (int i = tid; i < TILE_NODES * 16; i += 256) {
            int r = i >> 4;                       // node row (16 chunks/row)
            if (nodeBase + r < N) {
                unsigned long long gaddr = (unsigned long long)(size_t)(srcA + i * 4);
                unsigned laddr = ldsABase + (unsigned)i * 16u;
                asm volatile("global_load_async_to_lds_b128 %0, %1, off"
                             :: "v"(laddr), "v"(gaddr) : "memory");
            } else {
                *(float4*)&ldsA32[i * 4] = make_float4(0.f, 0.f, 0.f, 0.f);
            }
        }

        // --- Stage W slice transposed: Wk[a*64+kk][n] -> ldsW[n*64+kk] ---
        const float* srcW = Wk + (size_t)a * 64 * OUT_SIZE;
        for (int i = tid; i < 64 * 64; i += 256) {
            int kk = i >> 6, n = i & 63;
            ldsW[n * 64 + kk] = (_Float16)srcW[i];
        }

        asm volatile("s_wait_asynccnt 0x0" ::: "memory");  // our async copies landed
        __syncthreads();                                   // all waves' copies landed

#pragma unroll
        for (int half = 0; half < 2; ++half) {
            const int cbase = half * 32;
            // A fragment: two contiguous 8-float runs from LDS (2x32B),
            // converted to f16 in registers (packed v_cvt).
            const float* ap = &ldsA32[(wave * 16 + m) * 64 + cbase + deltaA];
            v8f alo32 = *(const v8f*)ap;
            v8f ahi32 = *(const v8f*)(ap + 8);
            v8h alo = __builtin_convertvector(alo32, v8h);
            v8h ahi = __builtin_convertvector(ahi32, v8h);
            v16h afrag = __builtin_shufflevector(alo, ahi,
                0, 1, 2, 3, 4, 5, 6, 7, 8, 9, 10, 11, 12, 13, 14, 15);
#pragma unroll
            for (int t = 0; t < 4; ++t) {
                // B fragment: contiguous 16 halfs (32B, aligned).
                const _Float16* bp = &ldsW[(t * 16 + m) * 64 + cbase + deltaB];
                v16h bfrag = *(const v16h*)bp;
                acc[t] = __builtin_amdgcn_wmma_f32_16x16x32_f16(
                    /*neg_a=*/false, afrag, /*neg_b=*/false, bfrag,
                    /*c_mod=*/(short)0, acc[t],
                    /*reuse_a=*/false, /*reuse_b=*/false);
            }
        }
    }

    // Epilogue. C/D layout: VGPR v, lanes 0-15 -> M=v, N=lane;
    //                                lanes 16-31 -> M=v+8, N=lane-16.
    const int rowOff = (lane < 16) ? 0 : 8;
#pragma unroll
    for (int t = 0; t < 4; ++t) {
        const int col = t * 16 + m;
#pragma unroll
        for (int v = 0; v < 8; ++v) {
            int node = nodeBase + wave * 16 + v + rowOff;
            if (node < N) {
                size_t o = (size_t)node * OUT_SIZE + col;
                float val = acc[t][v];
                if (initOut) out[o] = val + bias[col];
                else         out[o] = out[o] + val;
            }
        }
    }
}

extern "C" void kernel_launch(void* const* d_in, const int* in_sizes, int n_in,
                              void* d_out, int out_size, void* d_ws, size_t ws_size,
                              hipStream_t stream) {
    const float* x       = (const float*)d_in[0];
    const float* ls_vals = (const float*)d_in[1];
    const float* weight  = (const float*)d_in[2];  // [K][512][64]
    const float* bias    = (const float*)d_in[3];
    const int*   ls_rows = (const int*)d_in[4];
    const int*   ls_cols = (const int*)d_in[5];
    float*       out     = (float*)d_out;

    const int N = in_sizes[0] / IN_SIZE;                 // 20000
    const int M = N_ANGLES * N;                          // 160000
    const int E = in_sizes[1];                           // 2.56M
    const int K = in_sizes[2] / (KDIM * OUT_SIZE);       // 8

    const size_t tBytes = (size_t)M * 64 * sizeof(float);
    float* Ta = (float*)d_ws;                            // ping
    float* Tb = (float*)((char*)d_ws + tBytes);          // pong

    const int mv4        = M * 16;                       // float4 count of a T buffer
    const int copyBlocks = (mv4 + 255) / 256;
    const int gemmBlocks = (N + TILE_NODES - 1) / TILE_NODES;
    const int spmmBlocks = (int)(((long long)E * 32 + 255) / 256);
    dim3 b256(256);

    // k = 0 : T0 = tile(x) ; out = feat(T0) @ W0 + bias
    tile_kernel<<<copyBlocks, b256, 0, stream>>>((const float4*)x, (float4*)Ta, N, mv4);
    gemm_kernel<<<gemmBlocks, b256, 0, stream>>>(Ta, weight, bias, out, N, 1);

    // k = 1 : T1 = L @ T0
    scale_kernel<<<copyBlocks, b256, 0, stream>>>((float4*)Tb, 0.0f, mv4);
    spmm_kernel<<<spmmBlocks, b256, 0, stream>>>(ls_vals, ls_rows, ls_cols, Ta, Tb, 1.0f, E);
    gemm_kernel<<<gemmBlocks, b256, 0, stream>>>(Tb, weight + (size_t)1 * KDIM * OUT_SIZE,
                                                 bias, out, N, 0);

    // k = 2..K-1 : Ta <- 2 L Tb - Ta   (in-place: negate then accumulate)
    for (int k = 2; k < K; ++k) {
        scale_kernel<<<copyBlocks, b256, 0, stream>>>((float4*)Ta, -1.0f, mv4);
        spmm_kernel<<<spmmBlocks, b256, 0, stream>>>(ls_vals, ls_rows, ls_cols, Tb, Ta, 2.0f, E);
        gemm_kernel<<<gemmBlocks, b256, 0, stream>>>(Ta, weight + (size_t)k * KDIM * OUT_SIZE,
                                                     bias, out, N, 0);
        float* tmp = Ta; Ta = Tb; Tb = tmp;
    }
    (void)n_in; (void)out_size; (void)ws_size;
}
